// PhysicsInformedNN_34196529611438
// MI455X (gfx1250) — matrix operand order repaired and checked by reading, not compiled, not run on hardware
//
#include <hip/hip_runtime.h>
#include <hip/hip_bf16.h>
#include <math.h>

typedef __attribute__((ext_vector_type(16))) _Float16 v16h;
typedef __attribute__((ext_vector_type(8)))  _Float16 v8h;
typedef __attribute__((ext_vector_type(8)))  float    v8f;

#define PITCH 264              // halves per LDS activation row (16B-aligned rows)
#define WAVES_PER_BLOCK 2
#define BLOCK (WAVES_PER_BLOCK * 32)

// Packed weight layout in d_ws (halves):
//  L0 : Kpad=32  (nkc=1),  Npad=256 (nnt=16) -> 16 frags * 512 = 8192
//  L1-5: Kpad=256 (nkc=8), Npad=256 (nnt=16) -> 128 frags * 512 = 65536 each
//  L6 : Kpad=256 (nkc=8),  Npad=16  (nnt=1)  -> 8 frags * 512   = 4096
#define WP_OFF_L6 335872       // 8192 + 5*65536

struct BiasPack { const float* p[7]; };

// ---------------------------------------------------------------------------
// Hardware tanh (CDNA5 TRANS op).
// ---------------------------------------------------------------------------
__device__ __forceinline__ float fast_tanh(float x)
{
#if __has_builtin(__builtin_amdgcn_tanhf)
    return __builtin_amdgcn_tanhf(x);
#else
    float r;
    asm("v_tanh_f32 %0, %1\n\tv_nop" : "=v"(r) : "v"(x));
    return r;
#endif
}

// ---------------------------------------------------------------------------
// Repack fp32 KxN weights into WMMA B-fragment order, f16.
// B 32x16 f16 layout: lane L holds N = L&15; half i holds K = kbase + (L>=16?8:0) + (i&7) + (i>=8?16:0).
// ---------------------------------------------------------------------------
__global__ void repack_kernel(const float* __restrict__ W, int K, int N,
                              int nkc, _Float16* __restrict__ out, int total)
{
    int idx = blockIdx.x * blockDim.x + threadIdx.x;
    if (idx >= total) return;
    int i    = idx & 15;
    int lane = (idx >> 4) & 31;
    int frag = idx >> 9;
    int kc   = frag % nkc;
    int nt   = frag / nkc;
    int k = kc * 32 + ((lane >= 16) ? 8 : 0) + (i & 7) + ((i >= 8) ? 16 : 0);
    int n = nt * 16 + (lane & 15);
    float v = (k < K && n < N) ? W[k * N + n] : 0.0f;
    out[idx] = (_Float16)v;
}

__global__ void zero_kernel(float* out) { if (threadIdx.x == 0) out[0] = 0.0f; }

// ---------------------------------------------------------------------------
// A 16x32 f16 fragment from the LDS row-major tile.
// lane L: row m = mbase + (L&15); halves 0..7 -> K = kc*32+sel8+i ; 8..15 -> +16.
// ---------------------------------------------------------------------------
__device__ __forceinline__ v16h load_afrag(const _Float16* H, int m, int sel8, int kc)
{
    const _Float16* p = H + m * PITCH + kc * 32 + sel8;
    v8h lo = *(const v8h*)p;
    v8h hi = *(const v8h*)(p + 16);
    v16h a;
#pragma unroll
    for (int i = 0; i < 8; ++i) { a[i] = lo[i]; a[8 + i] = hi[i]; }
    return a;
}

__device__ __forceinline__ v8f wmma_f16(v16h a, v16h b, v8f c)
{
    return __builtin_amdgcn_wmma_f32_16x16x32_f16(false, a, false, b,
                                                  (short)0, c, false, false);
}

// Epilogue: primal rows (j==0 or j==4 -> m%4==0) get tanh(acc+bias);
// tangent rows get acc*(1-h^2) and no bias. BC mode: every row primal.
__device__ __forceinline__ void store_act(_Float16* H, v8f c, int sel8, int nt, int n,
                                          float bn, bool combined)
{
    float deriv = 0.0f;
#pragma unroll
    for (int j = 0; j < 8; ++j) {
        float v = c[j], hv;
        if (!combined || (j & 3) == 0) {
            hv = fast_tanh(v + bn);
            deriv = 1.0f - hv * hv;
        } else {
            hv = v * deriv;
        }
        H[(sel8 + j) * PITCH + nt * 16 + n] = (_Float16)hv;
    }
}

// ---------------------------------------------------------------------------
// One dense layer over a 32-row tile (two 16-row WMMA tiles sharing B frags).
// Accumulation starts from WMMA's inline-0 C operand; bias is applied in the
// epilogue (fused into the tanh input for primal rows).
// ---------------------------------------------------------------------------
template <int NKC>
__device__ __forceinline__ void hidden_layer(_Float16* H,
                                             const _Float16* __restrict__ Wl,
                                             const float* __restrict__ bias,
                                             int lane, bool combined)
{
    const int mA   = lane & 15;
    const int sel8 = (lane >> 4) << 3;
    const int n    = lane & 15;

    v16h af0[NKC], af1[NKC];
#pragma unroll
    for (int kc = 0; kc < NKC; ++kc) {
        af0[kc] = load_afrag(H, mA, sel8, kc);
        af1[kc] = load_afrag(H, 16 + mA, sel8, kc);
    }
#pragma unroll
    for (int nt = 0; nt < 16; ++nt) {
        float bn = bias[nt * 16 + n];
        const v8f zero = {};
        v16h b = *(const v16h*)(Wl + (nt * NKC * 32 + lane) * 16);
        v8f c0 = wmma_f16(af0[0], b, zero);
        v8f c1 = wmma_f16(af1[0], b, zero);
#pragma unroll
        for (int kc = 1; kc < NKC; ++kc) {
            v16h bk = *(const v16h*)(Wl + ((nt * NKC + kc) * 32 + lane) * 16);
            c0 = wmma_f16(af0[kc], bk, c0);
            c1 = wmma_f16(af1[kc], bk, c1);
        }
        store_act(H, c0, sel8, nt, n, bn, combined);
        store_act(H + 16 * PITCH, c1, sel8, nt, n, bn, combined);
    }
}

// Final layer 256 -> 3 (N padded to 16), linear; writes f32 into Sf (32x16).
__device__ __forceinline__ void final_layer(const _Float16* H, float* Sf,
                                            const _Float16* __restrict__ Wl,
                                            const float* __restrict__ b6,
                                            int lane, bool combined)
{
    const int mA   = lane & 15;
    const int sel8 = (lane >> 4) << 3;
    const int n    = lane & 15;

    v16h af0[8], af1[8];
#pragma unroll
    for (int kc = 0; kc < 8; ++kc) {
        af0[kc] = load_afrag(H, mA, sel8, kc);
        af1[kc] = load_afrag(H, 16 + mA, sel8, kc);
    }
    float bn = (n < 3) ? b6[n] : 0.0f;
    const v8f zero = {};
    {
        v16h b = *(const v16h*)(Wl + (lane) * 16);
        v8f c0 = wmma_f16(af0[0], b, zero);
        v8f c1 = wmma_f16(af1[0], b, zero);
#pragma unroll
        for (int kc = 1; kc < 8; ++kc) {
            v16h bk = *(const v16h*)(Wl + (kc * 32 + lane) * 16);
            c0 = wmma_f16(af0[kc], bk, c0);
            c1 = wmma_f16(af1[kc], bk, c1);
        }
#pragma unroll
        for (int j = 0; j < 8; ++j) {
            float add = (!combined || (j & 3) == 0) ? bn : 0.0f;
            Sf[(sel8 + j) * 16 + n]      = c0[j] + add;
            Sf[(16 + sel8 + j) * 16 + n] = c1[j] + add;
        }
    }
}

__device__ __forceinline__ void run_mlp(_Float16* H, float* Sf,
                                        const _Float16* __restrict__ Wp,
                                        BiasPack bp, int lane, bool combined)
{
    hidden_layer<1>(H, Wp, bp.p[0], lane, combined);
    for (int l = 1; l <= 5; ++l)
        hidden_layer<8>(H, Wp + 8192 + (l - 1) * 65536, bp.p[l], lane, combined);
    final_layer(H, Sf, Wp + WP_OFF_L6, bp.p[6], lane, combined);
}

// ---------------------------------------------------------------------------
// Boundary-condition loss: 32 points per wave (all rows primal).
// ---------------------------------------------------------------------------
__global__ __launch_bounds__(BLOCK) void bc_kernel(
    const float* __restrict__ x0, const float* __restrict__ y0,
    const float* __restrict__ bx0, const float* __restrict__ by0,
    const float* __restrict__ bz0,
    const float* __restrict__ lb, const float* __restrict__ ub,
    const _Float16* __restrict__ Wp, BiasPack bp,
    float* __restrict__ out, int N0)
{
    __shared__ __align__(16) _Float16 Hs[WAVES_PER_BLOCK][32 * PITCH];
    __shared__ float Sfs[WAVES_PER_BLOCK][512];
    __shared__ float blk_acc;

    int lane  = threadIdx.x & 31;
    int wslot = threadIdx.x >> 5;
    int wave  = blockIdx.x * WAVES_PER_BLOCK + wslot;
    int pt0   = wave * 32;

    if (threadIdx.x == 0) blk_acc = 0.0f;
    __syncthreads();

    _Float16* H  = Hs[wslot];
    float*    Sf = Sfs[wslot];

#pragma unroll
    for (int r = 0; r < 32; ++r) {
        int k = lane;
        float v = 0.0f;
        if (k < 3) {
            float p = (k == 0) ? x0[pt0 + r] : (k == 1) ? y0[pt0 + r] : 0.0f;
            v = 2.0f * (p - lb[k]) / (ub[k] - lb[k]) - 1.0f;
        }
        H[r * PITCH + k] = (_Float16)v;
    }

    run_mlp(H, Sf, Wp, bp, lane, false);

    int p = pt0 + lane;
    float ex = Sf[lane * 16 + 0] - bx0[p];
    float ey = Sf[lane * 16 + 1] - by0[p];
    float ez = Sf[lane * 16 + 2] - bz0[p];
    float contrib = (ex * ex + ey * ey + ez * ez) / (float)N0;

    atomicAdd(&blk_acc, contrib);
    __syncthreads();
    if (threadIdx.x == 0) atomicAdd(out, blk_acc);
}

// ---------------------------------------------------------------------------
// Field loss: 8 points per wave; rows 4p+{0..3} = [primal, d/dx, d/dy, d/dz].
// ---------------------------------------------------------------------------
__global__ __launch_bounds__(BLOCK) void field_kernel(
    const float* __restrict__ xyz,
    const float* __restrict__ lb, const float* __restrict__ ub,
    const _Float16* __restrict__ Wp, BiasPack bp,
    float* __restrict__ out, int NF)
{
    __shared__ __align__(16) _Float16 Hs[WAVES_PER_BLOCK][32 * PITCH];
    __shared__ float Sfs[WAVES_PER_BLOCK][512];
    __shared__ float blk_acc;

    int lane  = threadIdx.x & 31;
    int wslot = threadIdx.x >> 5;
    int wave  = blockIdx.x * WAVES_PER_BLOCK + wslot;
    int pt0   = wave * 8;

    if (threadIdx.x == 0) blk_acc = 0.0f;
    __syncthreads();

    _Float16* H  = Hs[wslot];
    float*    Sf = Sfs[wslot];

#pragma unroll
    for (int r = 0; r < 32; ++r) {
        int k = lane;
        int p = pt0 + (r >> 2);
        int t = r & 3;
        float v = 0.0f;
        if (k < 3) {
            float inv = ub[k] - lb[k];
            if (t == 0)          v = 2.0f * (xyz[p * 3 + k] - lb[k]) / inv - 1.0f;
            else if (k == t - 1) v = 2.0f / inv;
        }
        H[r * PITCH + k] = (_Float16)v;
    }

    run_mlp(H, Sf, Wp, bp, lane, true);

    float contrib = 0.0f;
    if (lane < 8) {
        int base = lane * 4;
        float bx  = Sf[(base + 0) * 16 + 0], by  = Sf[(base + 0) * 16 + 1], bz  = Sf[(base + 0) * 16 + 2];
        float jxx = Sf[(base + 1) * 16 + 0], jxy = Sf[(base + 1) * 16 + 1], jxz = Sf[(base + 1) * 16 + 2];
        float jyx = Sf[(base + 2) * 16 + 0], jyy = Sf[(base + 2) * 16 + 1], jyz = Sf[(base + 2) * 16 + 2];
        float jzx = Sf[(base + 3) * 16 + 0], jzy = Sf[(base + 3) * 16 + 1], jzz = Sf[(base + 3) * 16 + 2];
        float dv = jxx + jyy + jzz;
        float jx = jyz - jzy, jy = jzx - jxz, jz = jxy - jyx;
        float c1 = jy * bz - jz * by;
        float c2 = jz * bx - jx * bz;
        float c3 = jx * by - jy * bx;
        contrib = (dv * dv + c1 * c1 + c2 * c2 + c3 * c3) / (float)NF;
    }
    atomicAdd(&blk_acc, contrib);
    __syncthreads();
    if (threadIdx.x == 0) atomicAdd(out, blk_acc);
}

// ---------------------------------------------------------------------------
extern "C" void kernel_launch(void* const* d_in, const int* in_sizes, int n_in,
                              void* d_out, int out_size, void* d_ws, size_t ws_size,
                              hipStream_t stream)
{
    const float* x0  = (const float*)d_in[0];
    const float* y0  = (const float*)d_in[1];
    const float* bx0 = (const float*)d_in[2];
    const float* by0 = (const float*)d_in[3];
    const float* bz0 = (const float*)d_in[4];
    const float* xyz = (const float*)d_in[5];
    const float* lb  = (const float*)d_in[6];
    const float* ub  = (const float*)d_in[7];

    const float* W[7];
    BiasPack bp;
    for (int l = 0; l < 7; ++l) {
        W[l]    = (const float*)d_in[8 + 2 * l];
        bp.p[l] = (const float*)d_in[9 + 2 * l];
    }

    _Float16* Wp = (_Float16*)d_ws;   // 339,968 halves = 664 KiB
    float* out = (float*)d_out;

    zero_kernel<<<1, 64, 0, stream>>>(out);

    repack_kernel<<<8192 / 256, 256, 0, stream>>>(W[0], 3, 256, 1, Wp, 8192);
    for (int l = 1; l <= 5; ++l)
        repack_kernel<<<65536 / 256, 256, 0, stream>>>(W[l], 256, 256, 8,
                                                       Wp + 8192 + (l - 1) * 65536, 65536);
    repack_kernel<<<4096 / 256, 256, 0, stream>>>(W[6], 256, 3, 8, Wp + WP_OFF_L6, 4096);

    int N0 = in_sizes[0];        // 16384
    int NF = in_sizes[5] / 3;    // 32768

    bc_kernel<<<N0 / 32 / WAVES_PER_BLOCK, BLOCK, 0, stream>>>(
        x0, y0, bx0, by0, bz0, lb, ub, Wp, bp, out, N0);
    field_kernel<<<NF / 8 / WAVES_PER_BLOCK, BLOCK, 0, stream>>>(
        xyz, lb, ub, Wp, bp, out, NF);
}
